// _DistributedQwen2MoeSparseMoeBlock_66589172957455
// MI455X (gfx1250) — matrix operand order repaired and checked by reading, not compiled, not run on hardware
//
#include <hip/hip_runtime.h>
#include <math.h>

#define T_TOK 2048
#define DIM   2048
#define NEXP  8
#define FFN   1408
#define TOPK  2

#define TM    32                 // token rows per workgroup tile (2 WMMA row-halves)
#define BLK   256                // 8 waves (wave32)
#define WAVES (BLK / 32)
#define FT_TILES (FFN / 16)      // 88 column tiles in phase 1
#define NT1 (FT_TILES / WAVES)   // 11 tiles per wave
#define DT_TILES (DIM / 16)      // 128 column tiles in phase 2

typedef __attribute__((ext_vector_type(16))) __bf16 v16bf;
typedef __attribute__((ext_vector_type(8)))  float  v8f;

// ---------------- Kernel 0: zero the output (harness poisons d_out) --------
__global__ void k_zero(float* __restrict__ out, int n) {
    int i = blockIdx.x * blockDim.x + threadIdx.x;
    if (i < n) out[i] = 0.0f;
}

// ---------------- Kernel 1: router (logits -> softmax -> top-2) ------------
__global__ void k_router(const float* __restrict__ x,
                         const float* __restrict__ gate_w,
                         int* __restrict__ topidx,
                         float* __restrict__ topw) {
    int t = blockIdx.x * blockDim.x + threadIdx.x;
    if (t >= T_TOK) return;

    float acc[NEXP];
#pragma unroll
    for (int e = 0; e < NEXP; ++e) acc[e] = 0.0f;

    const float* xr = x + (size_t)t * DIM;
    for (int d = 0; d < DIM; ++d) {
        float xv = xr[d];
#pragma unroll
        for (int e = 0; e < NEXP; ++e) acc[e] += xv * gate_w[e * DIM + d];
    }

    // softmax over 8 logits
    float m = acc[0];
#pragma unroll
    for (int e = 1; e < NEXP; ++e) m = fmaxf(m, acc[e]);
    float s = 0.0f;
#pragma unroll
    for (int e = 0; e < NEXP; ++e) { acc[e] = __expf(acc[e] - m); s += acc[e]; }
    float inv = __builtin_amdgcn_rcpf(s);
#pragma unroll
    for (int e = 0; e < NEXP; ++e) acc[e] *= inv;

    // top-2, ties -> lowest index (matches jax.lax.top_k)
    int i0 = 0; float v0 = acc[0];
#pragma unroll
    for (int e = 1; e < NEXP; ++e) if (acc[e] > v0) { v0 = acc[e]; i0 = e; }
    int i1 = -1; float v1 = -1.0f;
#pragma unroll
    for (int e = 0; e < NEXP; ++e) if (e != i0 && acc[e] > v1) { v1 = acc[e]; i1 = e; }

    topidx[t * TOPK + 0] = i0;  topw[t * TOPK + 0] = v0;
    topidx[t * TOPK + 1] = i1;  topw[t * TOPK + 1] = v1;
}

// ---------------- Kernel 2: deterministic per-expert compaction ------------
__global__ void k_compact(const int* __restrict__ topidx,
                          const float* __restrict__ topw,
                          int* __restrict__ tokens,
                          float* __restrict__ combw,
                          int* __restrict__ counts) {
    int e = threadIdx.x;
    if (e >= NEXP) return;
    int cnt = 0;
    for (int t = 0; t < T_TOK; ++t) {
#pragma unroll
        for (int k = 0; k < TOPK; ++k) {
            if (topidx[t * TOPK + k] == e) {
                tokens[e * T_TOK + cnt] = t;
                combw[e * T_TOK + cnt]  = topw[t * TOPK + k];
                ++cnt;
            }
        }
    }
    counts[e] = cnt;
}

// ---------------- Kernel 3: per-expert fused MLP via bf16 WMMA -------------
// Each block: expert = blockIdx.y, 32 gathered token rows = blockIdx.x*32.
// Phase 1: h = silu(x*Wg) * (x*Wu)  -> 32x1408 bf16 tile in LDS
// Phase 2: y = h * Wd, scaled by routing weight, atomic-added to out.
// 216 KB LDS/workgroup: only possible because CDNA5 gives 320 KB per WGP.
__global__ __launch_bounds__(BLK)
void k_expert(const float* __restrict__ x,
              const float* __restrict__ w_gate,
              const float* __restrict__ w_up,
              const float* __restrict__ w_down,
              const int* __restrict__ tokens,
              const float* __restrict__ combw,
              const int* __restrict__ counts,
              float* __restrict__ out) {
    __shared__ __bf16 xs[TM * DIM];   // 32 x 2048 bf16 = 128 KB
    __shared__ __bf16 hs[TM * FFN];   // 32 x 1408 bf16 =  88 KB

    const int e    = blockIdx.y;
    const int cnt  = counts[e];
    const int row0 = blockIdx.x * TM;
    if (row0 >= cnt) return;
    const int rows = min(TM, cnt - row0);

    const int tid  = threadIdx.x;
    const int wave = tid >> 5;
    const int lane = tid & 31;
    const int half = lane >> 4;      // 0: lanes 0-15, 1: lanes 16-31
    const int l16  = lane & 15;
    const int kbA  = half * 8;       // A-frag K interleave (ISA 7.12.2)
    const int kbB  = half * 16;      // prefetch K offset (B loads use kbA order)

    // ---- stage 32 gathered x rows into LDS as bf16 (pad rows -> zeros) ----
    for (int idx = tid; idx < TM * DIM; idx += BLK) {
        int r = idx >> 11;           // / DIM
        int c = idx & (DIM - 1);
        float v = 0.0f;
        if (r < rows) {
            int tok = tokens[e * T_TOK + row0 + r];
            v = x[(size_t)tok * DIM + c];
        }
        xs[r * DIM + c] = (__bf16)v;
    }
    __syncthreads();

    const float* wg = w_gate + (size_t)e * DIM * FFN;
    const float* wu = w_up   + (size_t)e * DIM * FFN;

    // ---- phase 1: 11 (16-col) tiles per wave across F; 2 row-halves ----
#pragma unroll 1
    for (int i = 0; i < NT1; ++i) {
        const int n0 = (wave + i * WAVES) * 16;
        v8f accg0 = {}, accu0 = {};
        v8f accg1 = {}, accu1 = {};
        for (int k0 = 0; k0 < DIM; k0 += 32) {
            v16bf a0, a1, bg, bu;
#pragma unroll
            for (int j = 0; j < 8; ++j) {
                a0[j]     = xs[l16 * DIM + k0 + kbA + j];
                a0[j + 8] = xs[l16 * DIM + k0 + 16 + kbA + j];
                a1[j]     = xs[(16 + l16) * DIM + k0 + kbA + j];
                a1[j + 8] = xs[(16 + l16) * DIM + k0 + 16 + kbA + j];
            }
#pragma unroll
            for (int j = 0; j < 16; ++j) {
                size_t kk = (size_t)(k0 + (j < 8 ? kbA + j : 16 + kbA + (j - 8)));
                bg[j] = (__bf16)wg[kk * FFN + n0 + l16];
                bu[j] = (__bf16)wu[kk * FFN + n0 + l16];
            }
            if (k0 + 32 < DIM) {   // prefetch next K-block (global_prefetch_b8)
                __builtin_prefetch(&wg[(size_t)(k0 + 32 + kbB) * FFN + n0 + l16], 0, 1);
                __builtin_prefetch(&wu[(size_t)(k0 + 32 + kbB) * FFN + n0 + l16], 0, 1);
            }
            // one B fragment pair feeds 4 WMMAs (2 row-halves x {gate,up})
            accg0 = __builtin_amdgcn_wmma_f32_16x16x32_bf16(
                false, a0, false, bg, (short)0, accg0, false, false);
            accu0 = __builtin_amdgcn_wmma_f32_16x16x32_bf16(
                false, a0, false, bu, (short)0, accu0, false, false);
            accg1 = __builtin_amdgcn_wmma_f32_16x16x32_bf16(
                false, a1, false, bg, (short)0, accg1, false, false);
            accu1 = __builtin_amdgcn_wmma_f32_16x16x32_bf16(
                false, a1, false, bu, (short)0, accu1, false, false);
        }
        // silu(g)*u with v_rcp_f32 (no IEEE-division fixup chain)
#pragma unroll
        for (int r = 0; r < 8; ++r) {
            float g0 = accg0[r];
            float h0 = g0 * __builtin_amdgcn_rcpf(1.0f + __expf(-g0)) * accu0[r];
            hs[(r + 8 * half) * FFN + n0 + l16] = (__bf16)h0;
            float g1 = accg1[r];
            float h1 = g1 * __builtin_amdgcn_rcpf(1.0f + __expf(-g1)) * accu1[r];
            hs[(16 + r + 8 * half) * FFN + n0 + l16] = (__bf16)h1;
        }
    }
    __syncthreads();   // h tile visible to all waves

    // ---- phase 2: y = h * Wd, 16 (16-col) tiles per wave across D ----
    const float* wd = w_down + (size_t)e * FFN * DIM;
#pragma unroll 1
    for (int nt = wave; nt < DT_TILES; nt += WAVES) {
        const int n0 = nt * 16;
        v8f acc0 = {}, acc1 = {};
        for (int k0 = 0; k0 < FFN; k0 += 32) {
            v16bf a0, a1, b;
#pragma unroll
            for (int j = 0; j < 8; ++j) {
                a0[j]     = hs[l16 * FFN + k0 + kbA + j];
                a0[j + 8] = hs[l16 * FFN + k0 + 16 + kbA + j];
                a1[j]     = hs[(16 + l16) * FFN + k0 + kbA + j];
                a1[j + 8] = hs[(16 + l16) * FFN + k0 + 16 + kbA + j];
            }
#pragma unroll
            for (int j = 0; j < 16; ++j) {
                size_t kk = (size_t)(k0 + (j < 8 ? kbA + j : 16 + kbA + (j - 8)));
                b[j] = (__bf16)wd[kk * DIM + n0 + l16];
            }
            if (k0 + 32 < FFN) {
                __builtin_prefetch(&wd[(size_t)(k0 + 32 + kbB) * DIM + n0 + l16], 0, 1);
            }
            acc0 = __builtin_amdgcn_wmma_f32_16x16x32_bf16(
                false, a0, false, b, (short)0, acc0, false, false);
            acc1 = __builtin_amdgcn_wmma_f32_16x16x32_bf16(
                false, a1, false, b, (short)0, acc1, false, false);
        }
#pragma unroll
        for (int r = 0; r < 8; ++r) {
            int m0 = r + 8 * half;          // rows 0-15 subtile
            if (m0 < rows) {
                int   tok = tokens[e * T_TOK + row0 + m0];
                float cw  = combw[e * T_TOK + row0 + m0];
                atomicAdd(&out[(size_t)tok * DIM + n0 + l16], cw * acc0[r]);
            }
            int m1 = 16 + r + 8 * half;     // rows 16-31 subtile
            if (m1 < rows) {
                int   tok = tokens[e * T_TOK + row0 + m1];
                float cw  = combw[e * T_TOK + row0 + m1];
                atomicAdd(&out[(size_t)tok * DIM + n0 + l16], cw * acc1[r]);
            }
        }
    }
}

// ---------------- host launch ----------------
extern "C" void kernel_launch(void* const* d_in, const int* in_sizes, int n_in,
                              void* d_out, int out_size, void* d_ws, size_t ws_size,
                              hipStream_t stream) {
    const float* x      = (const float*)d_in[0];   // [1,2048,2048]
    const float* gate_w = (const float*)d_in[1];   // [8,2048]
    const float* w_gate = (const float*)d_in[2];   // [8,2048,1408]
    const float* w_up   = (const float*)d_in[3];   // [8,2048,1408]
    const float* w_down = (const float*)d_in[4];   // [8,1408,2048]
    float* out = (float*)d_out;                    // [1,2048,2048] f32

    // workspace layout (~164 KB)
    char* ws = (char*)d_ws;
    int*   topidx = (int*)ws;    ws += (size_t)T_TOK * TOPK * sizeof(int);
    float* topw   = (float*)ws;  ws += (size_t)T_TOK * TOPK * sizeof(float);
    int*   tokens = (int*)ws;    ws += (size_t)NEXP * T_TOK * sizeof(int);
    float* combw  = (float*)ws;  ws += (size_t)NEXP * T_TOK * sizeof(float);
    int*   counts = (int*)ws;

    const int n_out = T_TOK * DIM;
    k_zero<<<(n_out + BLK - 1) / BLK, BLK, 0, stream>>>(out, n_out);
    k_router<<<(T_TOK + BLK - 1) / BLK, BLK, 0, stream>>>(x, gate_w, topidx, topw);
    k_compact<<<1, 32, 0, stream>>>(topidx, topw, tokens, combw, counts);

    dim3 grid(T_TOK / TM, NEXP);   // 64 row tiles x 8 experts
    k_expert<<<grid, BLK, 0, stream>>>(x, w_gate, w_up, w_down,
                                       tokens, combw, counts, out);
}
